// GraphAttentionLayer_44023414784635
// MI455X (gfx1250) — matrix operand order repaired and checked by reading, not compiled, not run on hardware
//
#include <hip/hip_runtime.h>
#include <hip/hip_bf16.h>
#include <math.h>

// Problem constants (from reference)
#define NNODE   100000
#define DIM     128
#define HEADS   8
#define CDIM    16
#define NEDGE   800000
#define ETOT    (NEDGE + NNODE)      // edges + self loops
#define NTILES  (NNODE / 16)         // 6250 exactly
#define NEG_SLOPE 0.2f

typedef __attribute__((ext_vector_type(16))) _Float16 v16h;
typedef __attribute__((ext_vector_type(8)))  float    v8f;

// ---------------------------------------------------------------------------
// Kernel 1: h = x @ W via v_wmma_f32_16x16x32_f16 (hi/lo f16 split for ~fp32
// accuracy). The per-head attention scalars are fused into the matrix pass:
//   [a_src | a_dst] = x @ (W @ [A_src | A_dst]) = x @ Wa      (Wa: 128x16)
// so tile 8 of each node tile's column loop produces a_src/a_dst directly in
// the WMMA accumulator — no cross-lane reduction epilogue at all.
// Block = 128 threads = 4 waves; each wave owns one 16-node row tile.
// ---------------------------------------------------------------------------
__global__ __launch_bounds__(128) void gat_gemm_attn(
    const float* __restrict__ x, const float* __restrict__ W,
    const float* __restrict__ attS, const float* __restrict__ attD,
    float* __restrict__ hOut, float* __restrict__ aS, float* __restrict__ aD)
{
    // W^T staged as f16 hi/lo: WT[col][k] so B operand reads are contiguous.
    __shared__ __align__(32) _Float16 WT_hi[DIM * DIM];
    __shared__ __align__(32) _Float16 WT_lo[DIM * DIM];
    // Fused attention matrix Wa^T[j][k], j in [0,16): j<8 -> src head j,
    // j>=8 -> dst head j-8.
    __shared__ __align__(32) _Float16 WaT_hi[16 * DIM];
    __shared__ __align__(32) _Float16 WaT_lo[16 * DIM];

    const int tid = threadIdx.x;
    for (int idx = tid; idx < DIM * DIM; idx += 128) {
        const int col = idx & 127;      // coalesced global read (idx == k*128+col)
        const int k   = idx >> 7;
        const float w = W[idx];
        const _Float16 hi = (_Float16)w;
        WT_hi[col * DIM + k] = hi;
        WT_lo[col * DIM + k] = (_Float16)(w - (float)hi);
    }
    // Wa[k][j] = sum_c W[k][t*16+c] * att{S,D}[t][c],  t = j&7
    for (int idx = tid; idx < 16 * DIM; idx += 128) {
        const int j = idx >> 7;          // 0..15
        const int k = idx & 127;         // 0..127
        const int t = j & 7;
        const float* wrow = W + (size_t)k * DIM + t * 16;   // 16 contiguous floats
        const float* att  = (j < 8) ? (attS + t * 16) : (attD + t * 16);
        float acc = 0.0f;
#pragma unroll
        for (int c = 0; c < 16; ++c) acc += wrow[c] * att[c];
        const _Float16 hi = (_Float16)acc;
        WaT_hi[j * DIM + k] = hi;
        WaT_lo[j * DIM + k] = (_Float16)(acc - (float)hi);
    }
    __syncthreads();

    const int wave = tid >> 5;
    const int lane = tid & 31;
    const int tile = blockIdx.x * 4 + wave;
    if (tile >= NTILES) return;

    const int nodeBase = tile * 16;
    const int laneLo   = lane & 15;
    const int hiHalf   = lane >> 4;            // 0: lanes 0-15, 1: lanes 16-31
    const int row      = nodeBase + laneLo;    // A-matrix M index

    // Load A (16x128 fp32) into 4 K-chunks of 16x32, f16 hi/lo split.
    v16h ah[4], al[4];
#pragma unroll
    for (int ch = 0; ch < 4; ++ch) {
        const int ka = ch * 32 + hiHalf * 16;  // contiguous K order, matches B
        const float* xr = x + (size_t)row * DIM + ka;
        const float4 f0 = *(const float4*)(xr +  0);
        const float4 f1 = *(const float4*)(xr +  4);
        const float4 f2 = *(const float4*)(xr +  8);
        const float4 f3 = *(const float4*)(xr + 12);
        float tmp[16] = { f0.x,f0.y,f0.z,f0.w, f1.x,f1.y,f1.z,f1.w,
                          f2.x,f2.y,f2.z,f2.w, f3.x,f3.y,f3.z,f3.w };
#pragma unroll
        for (int i = 0; i < 16; ++i) {
            const _Float16 hh = (_Float16)tmp[i];
            ah[ch][i] = hh;
            al[ch][i] = (_Float16)(tmp[i] - (float)hh);
        }
    }

    // Tiles 0..7: h columns (one head each).  Tile 8: fused [a_src|a_dst].
#pragma unroll
    for (int t = 0; t < 9; ++t) {
        const _Float16* bhBase = (t < 8) ? &WT_hi[(t * 16 + laneLo) * DIM]
                                         : &WaT_hi[laneLo * DIM];
        const _Float16* blBase = (t < 8) ? &WT_lo[(t * 16 + laneLo) * DIM]
                                         : &WaT_lo[laneLo * DIM];
        v8f acc = {};
#pragma unroll
        for (int ch = 0; ch < 4; ++ch) {
            const int k0  = ch * 32 + hiHalf * 16;         // B layout K base
            const v16h bh = *(const v16h*)(bhBase + k0);
            const v16h bl = *(const v16h*)(blBase + k0);
            acc = __builtin_amdgcn_wmma_f32_16x16x32_f16(false, ah[ch], false, bh,
                                                         (short)0, acc, false, false);
            acc = __builtin_amdgcn_wmma_f32_16x16x32_f16(false, al[ch], false, bh,
                                                         (short)0, acc, false, false);
            acc = __builtin_amdgcn_wmma_f32_16x16x32_f16(false, ah[ch], false, bl,
                                                         (short)0, acc, false, false);
        }

        if (t < 8) {
#pragma unroll
            for (int r = 0; r < 8; ++r) {
                const int node = nodeBase + r + hiHalf * 8;    // D layout M index
                hOut[(size_t)node * DIM + t * 16 + laneLo] = acc[r];
            }
        } else {
#pragma unroll
            for (int r = 0; r < 8; ++r) {
                const int node = nodeBase + r + hiHalf * 8;
                if (laneLo < 8) aS[(size_t)node * HEADS + laneLo]       = acc[r];
                else            aD[(size_t)node * HEADS + (laneLo - 8)] = acc[r];
            }
        }
    }
}

// ---------------------------------------------------------------------------
// Kernel 2: init accumulators (d_out is poisoned by harness).
// ---------------------------------------------------------------------------
__global__ __launch_bounds__(256) void gat_init(
    float* __restrict__ outF, float* __restrict__ mx, float* __restrict__ denom)
{
    const size_t i = (size_t)blockIdx.x * 256 + threadIdx.x;
    if (i < (size_t)NNODE * DIM) outF[i] = 0.0f;
    if (i < (size_t)NNODE * HEADS) { mx[i] = -INFINITY; denom[i] = 0.0f; }
}

__device__ __forceinline__ void atomicMaxF32(float* addr, float val) {
    if (val >= 0.0f) atomicMax((int*)addr, __float_as_int(val));
    else             atomicMin((unsigned int*)addr, __float_as_uint(val));
}

__device__ __forceinline__ void edgeEndpoints(const long long* __restrict__ ei,
                                              long long e, long long& s, long long& d) {
    if (e < NEDGE) { s = ei[e]; d = ei[NEDGE + e]; }
    else           { s = d = e - NEDGE; }                  // self loop
}

// ---------------------------------------------------------------------------
// Kernel 3: segment max of leaky_relu logits; also emits ei (with self loops).
// ---------------------------------------------------------------------------
__global__ __launch_bounds__(256) void gat_edge_max(
    const long long* __restrict__ ei, const float* __restrict__ aS,
    const float* __restrict__ aD, float* __restrict__ mx,
    long long* __restrict__ eiOut, int writeEi)
{
    const long long e = (long long)blockIdx.x * 256 + threadIdx.x;
    if (e >= ETOT) return;
    long long s, d; edgeEndpoints(ei, e, s, d);
    if (writeEi) { eiOut[e] = s; eiOut[(long long)ETOT + e] = d; }
    const float* ps = aS + s * HEADS;
    const float* pd = aD + d * HEADS;
#pragma unroll
    for (int h = 0; h < HEADS; ++h) {
        float v = ps[h] + pd[h];
        v = v > 0.0f ? v : NEG_SLOPE * v;
        atomicMaxF32(&mx[d * HEADS + h], v);
    }
}

// ---------------------------------------------------------------------------
// Kernel 4: denom = segment_sum(exp(logit - mx[dst]))
// ---------------------------------------------------------------------------
__global__ __launch_bounds__(256) void gat_edge_sum(
    const long long* __restrict__ ei, const float* __restrict__ aS,
    const float* __restrict__ aD, const float* __restrict__ mx,
    float* __restrict__ denom)
{
    const long long e = (long long)blockIdx.x * 256 + threadIdx.x;
    if (e >= ETOT) return;
    long long s, d; edgeEndpoints(ei, e, s, d);
    const float* ps = aS + s * HEADS;
    const float* pd = aD + d * HEADS;
#pragma unroll
    for (int h = 0; h < HEADS; ++h) {
        float v = ps[h] + pd[h];
        v = v > 0.0f ? v : NEG_SLOPE * v;
        const float ex = expf(v - mx[d * HEADS + h]);
        atomicAdd(&denom[d * HEADS + h], ex);
    }
}

// ---------------------------------------------------------------------------
// Kernel 5: alpha = ex/denom; out[dst] += alpha * h[src].
// One wave per edge; lane handles 4 columns (head = lane/4). out fits in L2,
// so the float atomics resolve L2-side.
// ---------------------------------------------------------------------------
__global__ __launch_bounds__(256) void gat_edge_scatter(
    const long long* __restrict__ ei, const float* __restrict__ hIn,
    const float* __restrict__ aS, const float* __restrict__ aD,
    const float* __restrict__ mx, const float* __restrict__ denom,
    float* __restrict__ outF, float* __restrict__ alphaOut, int writeAlpha)
{
    const int lane = threadIdx.x & 31;
    const long long e = (long long)blockIdx.x * 8 + (threadIdx.x >> 5);
    if (e >= ETOT) return;
    long long s, d; edgeEndpoints(ei, e, s, d);

    const int h = lane >> 2;                       // head for this lane's 4 cols
    float v = aS[s * HEADS + h] + aD[d * HEADS + h];
    v = v > 0.0f ? v : NEG_SLOPE * v;
    const float ex    = expf(v - mx[d * HEADS + h]);
    const float alpha = ex / (denom[d * HEADS + h] + 1e-16f);
    if (writeAlpha && (lane & 3) == 0) alphaOut[e * HEADS + h] = alpha;

    const int cbase = lane * 4;
    const float4 hv = *(const float4*)&hIn[(size_t)s * DIM + cbase];
    float* op = &outF[(size_t)d * DIM + cbase];
    atomicAdd(op + 0, hv.x * alpha);
    atomicAdd(op + 1, hv.y * alpha);
    atomicAdd(op + 2, hv.z * alpha);
    atomicAdd(op + 3, hv.w * alpha);
}

// ---------------------------------------------------------------------------
// Kernel 6: out += bias
// ---------------------------------------------------------------------------
__global__ __launch_bounds__(256) void gat_bias(
    float* __restrict__ outF, const float* __restrict__ bias)
{
    const size_t i = (size_t)blockIdx.x * 256 + threadIdx.x;
    if (i < (size_t)NNODE * DIM) outF[i] += bias[i & (DIM - 1)];
}

// ---------------------------------------------------------------------------
extern "C" void kernel_launch(void* const* d_in, const int* in_sizes, int n_in,
                              void* d_out, int out_size, void* d_ws, size_t ws_size,
                              hipStream_t stream)
{
    const float*     x    = (const float*)d_in[0];
    const long long* ei   = (const long long*)d_in[1];   // int64 in reference
    const float*     W    = (const float*)d_in[2];
    const float*     attS = (const float*)d_in[3];
    const float*     attD = (const float*)d_in[4];
    const float*     bias = (const float*)d_in[5];

    float* outF = (float*)d_out;

    // Workspace layout (floats): h | a_src | a_dst | mx | denom  (~64 MB)
    float* h     = (float*)d_ws;
    float* aS    = h  + (size_t)NNODE * DIM;
    float* aD    = aS + (size_t)NNODE * HEADS;
    float* mx    = aD + (size_t)NNODE * HEADS;
    float* denom = mx + (size_t)NNODE * HEADS;

    // Tuple outputs (out, ei int64, alpha) concatenated; only write extras if
    // d_out is actually sized for the full tuple.
    const size_t outFloats   = (size_t)NNODE * DIM;
    const size_t eiBytes     = (size_t)ETOT * 2 * sizeof(long long);
    const size_t alphaBytes  = (size_t)ETOT * HEADS * sizeof(float);
    const size_t totalBytes  = outFloats * 4 + eiBytes + alphaBytes;
    const int writeExtra = ((size_t)out_size * 4 >= totalBytes) ? 1 : 0;
    long long* eiOut    = (long long*)((char*)d_out + outFloats * 4);
    float*     alphaOut = (float*)((char*)d_out + outFloats * 4 + eiBytes);

    const int initBlocks = (int)(((size_t)NNODE * DIM + 255) / 256);
    gat_init<<<initBlocks, 256, 0, stream>>>(outF, mx, denom);

    gat_gemm_attn<<<(NTILES + 3) / 4, 128, 0, stream>>>(x, W, attS, attD, h, aS, aD);

    const int edgeBlocks = (ETOT + 255) / 256;
    gat_edge_max<<<edgeBlocks, 256, 0, stream>>>(ei, aS, aD, mx, eiOut, writeExtra);
    gat_edge_sum<<<edgeBlocks, 256, 0, stream>>>(ei, aS, aD, mx, denom);

    gat_edge_scatter<<<(ETOT + 7) / 8, 256, 0, stream>>>(
        ei, h, aS, aD, mx, denom, outF, alphaOut, writeExtra);

    gat_bias<<<initBlocks, 256, 0, stream>>>(outF, bias);
}
